// PointNetSkeleton_Concat_91096256348440
// MI455X (gfx1250) — matrix operand order repaired and checked by reading, not compiled, not run on hardware
//
#include <hip/hip_runtime.h>
#include <hip/hip_bf16.h>

typedef __attribute__((ext_vector_type(16))) __bf16 v16bf;
typedef __attribute__((ext_vector_type(8)))  __bf16 v8bf;
typedef __attribute__((ext_vector_type(8)))  float  v8f;
typedef __attribute__((ext_vector_type(4)))  unsigned int u32x4;
typedef __attribute__((ext_vector_type(8)))  int i32x8;
typedef __attribute__((ext_vector_type(4)))  int i32x4;

#define BSZ 4
#define PN  2048
#define KN  64
#define SS1 1024
#define SS2 256
#define FMAXV 3.402823466e+38f

// ---------------------------------------------------------------------------
// Farthest point sampling: one block per graph. dmin lives in LDS across steps.
// ---------------------------------------------------------------------------
__global__ void fps_kernel(const float* __restrict__ pos, int P, int S,
                           int* __restrict__ idx, float* __restrict__ qpos)
{
    __shared__ float dmin[2048];
    __shared__ float rv[256];
    __shared__ int   ri[256];
    __shared__ float lastp[3];
    const int b   = blockIdx.x;
    const int tid = threadIdx.x;
    const float* p = pos + (size_t)b * P * 3;

    for (int i = tid; i < P; i += 256) dmin[i] = FMAXV;
    if (tid == 0) {
        idx[b * S] = 0;
        qpos[((size_t)b * S) * 3 + 0] = p[0];
        qpos[((size_t)b * S) * 3 + 1] = p[1];
        qpos[((size_t)b * S) * 3 + 2] = p[2];
        lastp[0] = p[0]; lastp[1] = p[1]; lastp[2] = p[2];
    }
    __syncthreads();

    for (int s = 1; s < S; ++s) {
        const float lx = lastp[0], ly = lastp[1], lz = lastp[2];
        float bestv = -1.0f; int besti = 0x7FFFFFFF;
        for (int i = tid; i < P; i += 256) {
            float dx = p[i*3+0]-lx, dy = p[i*3+1]-ly, dz = p[i*3+2]-lz;
            float d  = dx*dx + dy*dy + dz*dz;
            float nd = fminf(dmin[i], d);
            dmin[i] = nd;
            if (nd > bestv) { bestv = nd; besti = i; }
        }
        rv[tid] = bestv; ri[tid] = besti;
        __syncthreads();
        for (int off = 128; off > 0; off >>= 1) {
            if (tid < off) {
                float ov = rv[tid + off]; int oi = ri[tid + off];
                if (ov > rv[tid] || (ov == rv[tid] && oi < ri[tid])) { rv[tid] = ov; ri[tid] = oi; }
            }
            __syncthreads();
        }
        if (tid == 0) {
            int j = ri[0];
            idx[b * S + s] = j;
            float jx = p[j*3+0], jy = p[j*3+1], jz = p[j*3+2];
            qpos[((size_t)b * S + s) * 3 + 0] = jx;
            qpos[((size_t)b * S + s) * 3 + 1] = jy;
            qpos[((size_t)b * S + s) * 3 + 2] = jz;
            lastp[0] = jx; lastp[1] = jy; lastp[2] = jz;
        }
        __syncthreads();
    }
}

// ---------------------------------------------------------------------------
// Radius neighbors: one thread per center, sequential scan keeps first-K order.
// ---------------------------------------------------------------------------
__global__ void radius_kernel(const float* __restrict__ qpos, const float* __restrict__ pos,
                              int S, int P, float r2,
                              int* __restrict__ order, int* __restrict__ cnt)
{
    int t = blockIdx.x * blockDim.x + threadIdx.x;
    if (t >= BSZ * S) return;
    int b = t / S;
    const float qx = qpos[t*3+0], qy = qpos[t*3+1], qz = qpos[t*3+2];
    const float* p = pos + (size_t)b * P * 3;
    int found = 0;
    for (int i = 0; i < P; ++i) {
        float dx = p[i*3+0]-qx, dy = p[i*3+1]-qy, dz = p[i*3+2]-qz;
        float d2 = dx*dx + dy*dy + dz*dz;
        if (d2 <= r2) { if (found < KN) order[(size_t)t*KN + found] = i; ++found; }
    }
    int c = found < KN ? found : KN;
    cnt[t] = c;
    for (int k = c; k < KN; ++k) order[(size_t)t*KN + k] = 0;
}

// SA1 features (bf16): rel = nbr_pos - qpos (3 ch); invalid slots zeroed.
__global__ void feat1_kernel(const float* __restrict__ pos, const float* __restrict__ qpos,
                             const int* __restrict__ order, const int* __restrict__ cnt,
                             int S, int P, __bf16* __restrict__ feat)
{
    int t = blockIdx.x * blockDim.x + threadIdx.x;
    if (t >= BSZ * S * KN) return;
    int center = t / KN, k = t % KN, b = center / S;
    float fx = 0.f, fy = 0.f, fz = 0.f;
    if (k < cnt[center]) {
        int pi = order[(size_t)t];
        const float* pp = pos + ((size_t)b * P + pi) * 3;
        fx = pp[0] - qpos[(size_t)center*3+0];
        fy = pp[1] - qpos[(size_t)center*3+1];
        fz = pp[2] - qpos[(size_t)center*3+2];
    }
    feat[(size_t)t*3+0] = (__bf16)fx;
    feat[(size_t)t*3+1] = (__bf16)fy;
    feat[(size_t)t*3+2] = (__bf16)fz;
}

// SA2 features (bf16): [x1(256) | rel(3)] per (center,k) row.
__global__ void feat2_kernel(const __bf16* __restrict__ x1, const float* __restrict__ qpos1,
                             const float* __restrict__ qpos2,
                             const int* __restrict__ order, const int* __restrict__ cnt,
                             __bf16* __restrict__ feat)
{
    long long e = (long long)blockIdx.x * blockDim.x + threadIdx.x;
    long long total = (long long)BSZ * SS2 * KN * 259;
    if (e >= total) return;
    int c = (int)(e % 259);
    long long row = e / 259;
    int center = (int)(row / KN), k = (int)(row % KN), b = center / SS2;
    float v = 0.f;
    if (k < cnt[center]) {
        int pi = order[row];
        if (c < 256) v = (float)x1[((size_t)b * SS1 + pi) * 256 + c];
        else v = qpos1[((size_t)b * SS1 + pi) * 3 + (c - 256)] - qpos2[(size_t)center * 3 + (c - 256)];
    }
    feat[e] = (__bf16)v;
}

// nn3 input (bf16): [x2(256) | qpos2(3)] per center row.
__global__ void feat3_kernel(const __bf16* __restrict__ x2, const float* __restrict__ qpos2,
                             __bf16* __restrict__ feat)
{
    int e = blockIdx.x * blockDim.x + threadIdx.x;
    if (e >= BSZ * SS2 * 259) return;
    int row = e / 259, c = e % 259;
    feat[e] = (c < 256) ? x2[(size_t)row * 256 + c] : (__bf16)qpos2[(size_t)row * 3 + (c - 256)];
}

// Masked max over K neighbors on bf16 activations; 0 when cnt==0.
__global__ void sa_pool_kernel(const __bf16* __restrict__ h, const int* __restrict__ cnt,
                               int rows, int C, __bf16* __restrict__ out)
{
    long long e = (long long)blockIdx.x * blockDim.x + threadIdx.x;
    if (e >= (long long)rows * C) return;
    int row = (int)(e / C), c = (int)(e % C);
    int n = cnt[row];
    float m = 0.f;
    if (n > 0) {
        m = -FMAXV;
        for (int k = 0; k < n; ++k) m = fmaxf(m, (float)h[((size_t)row * KN + k) * C + c]);
    }
    out[e] = (__bf16)m;
}

__global__ void gpool_kernel(const __bf16* __restrict__ in, int S, int C, __bf16* __restrict__ out)
{
    int e = blockIdx.x * blockDim.x + threadIdx.x;
    if (e >= BSZ * C) return;
    int b = e / C, c = e % C;
    float m = -FMAXV;
    for (int s = 0; s < S; ++s) m = fmaxf(m, (float)in[((size_t)b * S + s) * C + c]);
    out[e] = (__bf16)m;
}

// ---------------------------------------------------------------------------
// bf16 WMMA GEMM with TDM-staged A tiles:
//   C = opt_relu(A[M,K](bf16) @ W[K,N](f32->bf16) + bias)
// 256 thr = 8 waves (4x2); block tile 128x64; wave tile 32x32 -> 4 WMMA/chunk.
// Wave 0 drives the Tensor Data Mover for the A tile (OOB reads return zero,
// so edge tiles need no special casing); all waves stage W transposed.
// ---------------------------------------------------------------------------
__global__ void gemm_bf16_wmma(const __bf16* __restrict__ A, const float* __restrict__ W,
                               const float* __restrict__ bias,
                               __bf16* __restrict__ Cb, float* __restrict__ Cf,
                               int M, int K, int N, int relu)
{
    __shared__ __bf16 sA[128][32];  // [m][k]  (filled by TDM, row-major contiguous)
    __shared__ __bf16 sB[64][32];   // [n][k]  (W transposed)
    const int tid  = threadIdx.x;
    const int lane = tid & 31;
    const int wave = tid >> 5;
    const int waveM = wave >> 1, waveN = wave & 1;   // 4 x 2
    const int bm0 = blockIdx.x * 128;
    const int bn0 = blockIdx.y * 64;

    v8f acc00 = {}, acc01 = {}, acc10 = {}, acc11 = {};

    const int wn  = tid & 63;          // W col (n)   0..63
    const int wk0 = (tid >> 6) * 8;    // W row base  0,8,16,24

    const unsigned ldsOff = (unsigned)(size_t)(&sA[0][0]);
    const bool fullN = (bn0 + 64 <= N);              // block-uniform

    for (int kc = 0; kc < K; kc += 32) {
        // ---- wave 0: issue TDM load of A tile [128 x 32] bf16 into sA ----
        if (wave == 0) {
            unsigned long long ga = (unsigned long long)(size_t)(A + (size_t)bm0 * K + kc);
            unsigned td0 = (unsigned)(K - kc);          // remaining cols (OOB -> 0)
            unsigned td1 = (unsigned)(M - bm0);         // remaining rows (OOB -> 0)
            unsigned long long st0 = (unsigned long long)(unsigned)K;   // row stride, elements

            u32x4 g0;
            g0[0] = 1u;                                               // count=1
            g0[1] = ldsOff;                                           // lds_addr
            g0[2] = (unsigned)(ga & 0xFFFFFFFFull);                   // global_addr lo
            g0[3] = (unsigned)((ga >> 32) & 0x01FFFFFFull) | (2u << 30); // addr hi | type=2

            i32x8 g1;
            g1[0] = (int)(1u << 16);                                  // data_size=2B
            g1[1] = (int)((td0 & 0xFFFFu) << 16);                     // tensor_dim0 lo
            g1[2] = (int)(((td0 >> 16) & 0xFFFFu) | ((td1 & 0xFFFFu) << 16));
            g1[3] = (int)(((td1 >> 16) & 0xFFFFu) | (32u << 16));     // tile_dim0=32
            g1[4] = (int)128;                                         // tile_dim1=128, tile_dim2=0
            g1[5] = (int)(st0 & 0xFFFFFFFFull);                       // dim0 stride lo
            g1[6] = (int)((st0 >> 32) & 0xFFFFull);                   // stride hi | dim1 stride=0
            g1[7] = 0;

            i32x4 gz4 = {0, 0, 0, 0};
            i32x8 gz8 = {0, 0, 0, 0, 0, 0, 0, 0};
            __builtin_amdgcn_tensor_load_to_lds(g0, g1, gz4, gz4, gz8, 0);
        }

        // ---- all waves: stage W tile transposed (64n x 32k bf16) ----
        // per-e load is wave-coalesced over n; LDS store is one contiguous b128
        {
            const int gn = bn0 + wn;
            if (kc + 32 < K)                               // uniform
                __builtin_prefetch(&W[(size_t)(kc + 32 + wk0) * N + (gn < N ? gn : N - 1)], 0, 1);
            v8bf p;
            if (fullN && kc + 32 <= K) {                   // uniform fast path: no clamps/selects
                const float* wp = W + (size_t)(kc + wk0) * N + gn;
                #pragma unroll
                for (int e = 0; e < 8; ++e) p[e] = (__bf16)wp[(size_t)e * N];
            } else {
                const int gnc = gn < N ? gn : N - 1;
                #pragma unroll
                for (int e = 0; e < 8; ++e) {
                    int gk  = kc + wk0 + e;
                    int gkc = gk < K ? gk : K - 1;
                    float v = W[(size_t)gkc * N + gnc];
                    p[e] = (__bf16)((gk < K && gn < N) ? v : 0.0f);
                }
            }
            *(v8bf*)&sB[wn][wk0] = p;
        }

        if (wave == 0) __builtin_amdgcn_s_wait_tensorcnt(0);
        __syncthreads();

        // ---- fragments ----
        const int mrow = waveM * 32 + (lane & 15);
        const int kh   = (lane >> 4) * 8;
        v16bf a0, a1;
        #pragma unroll
        for (int e = 0; e < 8; ++e) { a0[e] = sA[mrow][kh + e];      a0[8+e] = sA[mrow][16 + kh + e]; }
        #pragma unroll
        for (int e = 0; e < 8; ++e) { a1[e] = sA[mrow + 16][kh + e]; a1[8+e] = sA[mrow + 16][16 + kh + e]; }
        const int kb = (lane >> 4) * 8;
        const int n0 = waveN * 32 + (lane & 15);
        v16bf b0, b1;
        #pragma unroll
        for (int e = 0; e < 8; ++e) { b0[e] = sB[n0][kb + e];      b0[8+e] = sB[n0][16 + kb + e]; }
        #pragma unroll
        for (int e = 0; e < 8; ++e) { b1[e] = sB[n0 + 16][kb + e]; b1[8+e] = sB[n0 + 16][16 + kb + e]; }

        acc00 = __builtin_amdgcn_wmma_f32_16x16x32_bf16(false, a0, false, b0, (short)0, acc00, false, false);
        acc01 = __builtin_amdgcn_wmma_f32_16x16x32_bf16(false, a0, false, b1, (short)0, acc01, false, false);
        acc10 = __builtin_amdgcn_wmma_f32_16x16x32_bf16(false, a1, false, b0, (short)0, acc10, false, false);
        acc11 = __builtin_amdgcn_wmma_f32_16x16x32_bf16(false, a1, false, b1, (short)0, acc11, false, false);
        __syncthreads();   // wave's ds reads retired (s_wait_dscnt before wmma) -> safe vs next TDM
    }

    // ---- epilogue: lane L -> col L%16, VGPR r -> row r + 8*(L/16) ----
    const int gmA  = bm0 + waveM * 32 + 8 * (lane >> 4);
    const int col0 = bn0 + waveN * 32 + (lane & 15);
    const int col1 = col0 + 16;
    const float bias0 = (col0 < N) ? bias[col0] : 0.0f;
    const float bias1 = (col1 < N) ? bias[col1] : 0.0f;
    const bool hasB = (Cb != nullptr);
    const bool hasF = (Cf != nullptr);
    #pragma unroll
    for (int r = 0; r < 8; ++r) {
        int g0r = gmA + r, g1r = gmA + 16 + r;
        if (g0r < M && col0 < N) {
            float v = acc00[r] + bias0; if (relu) v = fmaxf(v, 0.f);
            if (hasB) Cb[(size_t)g0r * N + col0] = (__bf16)v;
            if (hasF) Cf[(size_t)g0r * N + col0] = v;
        }
        if (g0r < M && col1 < N) {
            float v = acc01[r] + bias1; if (relu) v = fmaxf(v, 0.f);
            if (hasB) Cb[(size_t)g0r * N + col1] = (__bf16)v;
            if (hasF) Cf[(size_t)g0r * N + col1] = v;
        }
        if (g1r < M && col0 < N) {
            float v = acc10[r] + bias0; if (relu) v = fmaxf(v, 0.f);
            if (hasB) Cb[(size_t)g1r * N + col0] = (__bf16)v;
            if (hasF) Cf[(size_t)g1r * N + col0] = v;
        }
        if (g1r < M && col1 < N) {
            float v = acc11[r] + bias1; if (relu) v = fmaxf(v, 0.f);
            if (hasB) Cb[(size_t)g1r * N + col1] = (__bf16)v;
            if (hasF) Cf[(size_t)g1r * N + col1] = v;
        }
    }
}

// ---------------------------------------------------------------------------

extern "C" void kernel_launch(void* const* d_in, const int* in_sizes, int n_in,
                              void* d_out, int out_size, void* d_ws, size_t ws_size,
                              hipStream_t stream) {
    (void)in_sizes; (void)n_in; (void)out_size; (void)ws_size;
    const float* pos = (const float*)d_in[0];
    // d_in[1] = batch (implicit, unused)
    const float* n1w0 = (const float*)d_in[2];  const float* n1b0 = (const float*)d_in[3];
    const float* n1w1 = (const float*)d_in[4];  const float* n1b1 = (const float*)d_in[5];
    const float* n1w2 = (const float*)d_in[6];  const float* n1b2 = (const float*)d_in[7];
    const float* n2w0 = (const float*)d_in[8];  const float* n2b0 = (const float*)d_in[9];
    const float* n2w1 = (const float*)d_in[10]; const float* n2b1 = (const float*)d_in[11];
    const float* n2w2 = (const float*)d_in[12]; const float* n2b2 = (const float*)d_in[13];
    const float* n3w0 = (const float*)d_in[14]; const float* n3b0 = (const float*)d_in[15];
    const float* n3w1 = (const float*)d_in[16]; const float* n3b1 = (const float*)d_in[17];
    const float* n3w2 = (const float*)d_in[18]; const float* n3b2 = (const float*)d_in[19];
    const float* hw0  = (const float*)d_in[20]; const float* hb0  = (const float*)d_in[21];
    const float* hw1  = (const float*)d_in[22]; const float* hb1  = (const float*)d_in[23];
    const float* hw2  = (const float*)d_in[24]; const float* hb2  = (const float*)d_in[25];
    const float* outw = (const float*)d_in[26]; const float* outb = (const float*)d_in[27];
    float* out = (float*)d_out;

    char* ws = (char*)d_ws;
    size_t off = 0;
    auto alloc = [&](size_t bytes) -> void* {
        off = (off + 255) & ~(size_t)255;
        void* p = ws + off;
        off += bytes;
        return p;
    };
    int*    idx1   = (int*)   alloc((size_t)BSZ*SS1*4);
    float*  qpos1  = (float*) alloc((size_t)BSZ*SS1*3*4);
    int*    idx2   = (int*)   alloc((size_t)BSZ*SS2*4);
    float*  qpos2  = (float*) alloc((size_t)BSZ*SS2*3*4);
    int*    order1 = (int*)   alloc((size_t)BSZ*SS1*KN*4);
    int*    cnt1   = (int*)   alloc((size_t)BSZ*SS1*4);
    int*    order2 = (int*)   alloc((size_t)BSZ*SS2*KN*4);
    int*    cnt2   = (int*)   alloc((size_t)BSZ*SS2*4);
    __bf16* x1     = (__bf16*)alloc((size_t)BSZ*SS1*256*2);
    __bf16* x2     = (__bf16*)alloc((size_t)BSZ*SS2*256*2);
    __bf16* g      = (__bf16*)alloc((size_t)BSZ*1024*2);
    __bf16* repBf  = (__bf16*)alloc((size_t)BSZ*512*2);
    __bf16* bufA   = (__bf16*)alloc((size_t)BSZ*SS1*KN*256*2);
    __bf16* bufB   = (__bf16*)alloc((size_t)BSZ*SS1*KN*256*2);

    auto gemm = [&](const __bf16* A, const float* W, const float* bias,
                    __bf16* Cb, float* Cf, int M, int K, int N, int relu) {
        dim3 grid((M + 127) / 128, (N + 63) / 64);
        gemm_bf16_wmma<<<grid, dim3(256), 0, stream>>>(A, W, bias, Cb, Cf, M, K, N, relu);
    };

    // ---- SA1 ----
    fps_kernel<<<BSZ, 256, 0, stream>>>(pos, PN, SS1, idx1, qpos1);
    radius_kernel<<<(BSZ*SS1 + 255) / 256, 256, 0, stream>>>(qpos1, pos, SS1, PN, 0.2f*0.2f, order1, cnt1);
    feat1_kernel<<<(BSZ*SS1*KN + 255) / 256, 256, 0, stream>>>(pos, qpos1, order1, cnt1, SS1, PN, bufA);
    const int R1 = BSZ*SS1*KN;                                // 262144 rows
    gemm(bufA, n1w0, n1b0, bufB, nullptr, R1,   3,  64, 1);
    gemm(bufB, n1w1, n1b1, bufA, nullptr, R1,  64,  64, 1);
    gemm(bufA, n1w2, n1b2, bufB, nullptr, R1,  64, 256, 0);
    sa_pool_kernel<<<(int)(((long long)BSZ*SS1*256 + 255) / 256), 256, 0, stream>>>(bufB, cnt1, BSZ*SS1, 256, x1);

    // ---- SA2 ----
    fps_kernel<<<BSZ, 256, 0, stream>>>(qpos1, SS1, SS2, idx2, qpos2);
    radius_kernel<<<(BSZ*SS2 + 255) / 256, 256, 0, stream>>>(qpos2, qpos1, SS2, SS1, 0.4f*0.4f, order2, cnt2);
    {
        long long tot = (long long)BSZ*SS2*KN*259;
        feat2_kernel<<<(int)((tot + 255) / 256), 256, 0, stream>>>(x1, qpos1, qpos2, order2, cnt2, bufA);
    }
    const int R2 = BSZ*SS2*KN;                                // 65536 rows
    gemm(bufA, n2w0, n2b0, bufB, nullptr, R2, 259, 256, 1);
    gemm(bufB, n2w1, n2b1, bufA, nullptr, R2, 256, 256, 1);
    gemm(bufA, n2w2, n2b2, bufB, nullptr, R2, 256, 256, 0);
    sa_pool_kernel<<<(int)(((long long)BSZ*SS2*256 + 255) / 256), 256, 0, stream>>>(bufB, cnt2, BSZ*SS2, 256, x2);

    // ---- nn3 + global max pool ----
    feat3_kernel<<<(BSZ*SS2*259 + 255) / 256, 256, 0, stream>>>(x2, qpos2, bufA);
    const int R3 = BSZ*SS2;                                   // 1024 rows
    gemm(bufA, n3w0, n3b0, bufB, nullptr, R3, 259,  256, 1);
    gemm(bufB, n3w1, n3b1, bufA, nullptr, R3, 256,  512, 1);
    gemm(bufA, n3w2, n3b2, bufB, nullptr, R3, 512, 1024, 0);
    gpool_kernel<<<(BSZ*1024 + 255) / 256, 256, 0, stream>>>(bufB, SS2, 1024, g);

    // ---- head ----
    gemm(g,    hw0, hb0, bufA,  nullptr, BSZ, 1024, 682, 1);
    gemm(bufA, hw1, hb1, bufB,  nullptr, BSZ,  682, 853, 1);
    gemm(bufB, hw2, hb2, repBf, out,     BSZ,  853, 512, 0);           // rep -> d_out[0:B*512)
    gemm(repBf, outw, outb, nullptr, out + (size_t)BSZ*512, BSZ, 512, 40, 0); // y
}
